// BDH_GPU_3513283248750
// MI455X (gfx1250) — matrix operand order repaired and checked by reading, not compiled, not run on hardware
//
#include <hip/hip_runtime.h>
#include <hip/hip_bf16.h>

// ---------------------------------------------------------------------------
// BDH forward pass for MI455X (gfx1250), wave32 + WMMA bf16 + TDM async load.
// All matmuls use v_wmma_f32_16x16x32_bf16 (bf16 in, f32 accumulate).
// Attention Q/K tiles staged via tensor_load_to_lds (TDM) with HW padding
// that reproduces the pitch-72 LDS layout the WMMA fragment loaders expect;
// the K tile is double-buffered and pipelined with s_wait_tensorcnt 1
// (tensor ops complete in-order per wave, ISA ch.7.1).
// ---------------------------------------------------------------------------

typedef __bf16 bf16;
typedef __attribute__((ext_vector_type(16))) __bf16 v16bf;
typedef __attribute__((ext_vector_type(8)))  float  v8f;
typedef __attribute__((ext_vector_type(4)))  unsigned int v4u;
typedef __attribute__((ext_vector_type(8)))  int v8i;
typedef __attribute__((ext_vector_type(4)))  int v4i;

constexpr int D_  = 256;    // model dim
constexpr int H_  = 4;      // heads
constexpr int NH_ = 8192;   // N / H
constexpr int N_  = 32768;  // latent dim
constexpr int B_  = 2;
constexpr int T_  = 2048;
constexpr int V_  = 256;    // vocab
constexpr int BT  = B_ * T_;        // 4096
constexpr int BHT = B_ * H_ * T_;   // 16384
constexpr int LAYERS = 6;

#define DEV static __device__ __forceinline__

union FragU { v16bf v; float4 q[2]; };

// A-fragment (M x 32 K-chunk) from row-major LDS tile, pitch `ld` elements.
// lane L: m = m0 + (L&15); element e -> k = (e&7) + 8*(L>>4) + 16*(e>>3).
DEV v16bf frag_a(const bf16* lds, int ld, int m0, int k0, int lane) {
  const bf16* p = lds + (m0 + (lane & 15)) * ld + k0 + ((lane >> 4) << 3);
  FragU f;
  f.q[0] = *(const float4*)(p);
  f.q[1] = *(const float4*)(p + 16);
  return f.v;
}

// B-fragment (32 K-chunk x N) from LDS tile stored TRANSPOSED [n][k], pitch ld.
// lane L: n = n0 + (L&15); element e -> k = (e&7) + 8*(L>>4) + 16*(e>>3).
DEV v16bf frag_b(const bf16* lds, int ld, int n0, int k0, int lane) {
  const bf16* p = lds + (n0 + (lane & 15)) * ld + k0 + ((lane >> 4) << 3);
  FragU f;
  f.q[0] = *(const float4*)(p);
  f.q[1] = *(const float4*)(p + 16);
  return f.v;
}

DEV v8f wmma_bf(v16bf a, v16bf b, v8f c) {
  return __builtin_amdgcn_wmma_f32_16x16x32_bf16(false, a, false, b, (short)0, c,
                                                 false, false);
}

DEV float wsum(float v) {
  for (int m = 16; m >= 1; m >>= 1) v += __shfl_xor(v, m, 32);
  return v;
}

// ---------------------------------------------------------------------------
// TDM: async 2D tile load global -> LDS.  Tile is `rows` x 64 bf16 elements,
// global row stride NH_ elements.  HW pad inserts 16 bytes after every
// 128-byte row (pad_interval code 4 = 32 DWORDs, pad_amount code 3 = 4
// DWORDs), producing LDS pitch 72 elements (144 B) -- the same layout
// frag_a/frag_b consume.  D# per CDNA5 ISA ch.8.
// ---------------------------------------------------------------------------
DEV void tdm_load_64w(unsigned lds_off, const bf16* gsrc, int rows) {
  unsigned long long ga = (unsigned long long)(const void*)gsrc;
  v4u g0;
  g0.x = 1u;                                   // count = 1, user mode
  g0.y = lds_off;                              // lds_addr (bytes)
  g0.z = (unsigned)ga;                         // global_addr[31:0]
  g0.w = (unsigned)((ga >> 32) & 0x01FFFFFFu) | (2u << 30);  // addr[56:32]|type=2
  const unsigned long long td0 = 0x7FFFFFFFull;   // tensor_dim0 (no OOB clip)
  const unsigned long long td1 = 0x7FFFFFFFull;   // tensor_dim1
  unsigned long long st0 = (unsigned long long)NH_;  // tensor_dim0_stride (elems)
  unsigned long long q0 = (1ull << 16)         // data_size = 1 -> 2 bytes
                        | (1ull << 20)         // pad_enable
                        | (4ull << 22)         // pad_interval: 32 DWORDs
                        | (3ull << 25)         // pad_amount: 4 DWORDs
                        | ((td0 & 0xFFFFull) << 48);
  unsigned long long q1 = ((td0 >> 16) & 0xFFFFull)
                        | ((td1 & 0xFFFFFFFFull) << 16)
                        | (64ull << 48);       // tile_dim0 = 64 elements
  unsigned long long q2 = (unsigned long long)(unsigned)rows   // tile_dim1
                        | ((st0 & 0xFFFFFFFFull) << 32);
  unsigned long long q3 = (st0 >> 32) & 0xFFFFull;
  v8i g1;
  g1[0] = (int)(unsigned)q0; g1[1] = (int)(q0 >> 32);
  g1[2] = (int)(unsigned)q1; g1[3] = (int)(q1 >> 32);
  g1[4] = (int)(unsigned)q2; g1[5] = (int)(q2 >> 32);
  g1[6] = (int)(unsigned)q3; g1[7] = (int)(q3 >> 32);
  v4i gz = {0, 0, 0, 0};
#if __clang_major__ >= 23
  v8i gz8 = {0, 0, 0, 0, 0, 0, 0, 0};
  __builtin_amdgcn_tensor_load_to_lds(g0, g1, gz, gz, gz8, 0);
#else
  __builtin_amdgcn_tensor_load_to_lds(g0, g1, gz, gz, 0);
#endif
}

// ---------------------------------------------------------------------------
// Weight conversion: decoder_x/y (H,D,NH) -> bf16 transposed (H,NH,D);
// encoder (N,D) -> bf16 transposed (D,N).
// ---------------------------------------------------------------------------
__global__ void k_convert(const float* __restrict__ dx, const float* __restrict__ dy,
                          const float* __restrict__ enc,
                          bf16* __restrict__ dxt, bf16* __restrict__ dyt,
                          bf16* __restrict__ enct) {
  long i0 = (long)blockIdx.x * blockDim.x + threadIdx.x;
  long stride = (long)gridDim.x * blockDim.x;
  const long ndx = (long)H_ * NH_ * D_;
  for (long i = i0; i < ndx; i += stride) {
    int d = (int)(i % D_);
    long hn = i / D_;
    int n = (int)(hn % NH_);
    int h = (int)(hn / NH_);
    long src = ((long)h * D_ + d) * NH_ + n;
    dxt[i] = (bf16)dx[src];
    dyt[i] = (bf16)dy[src];
  }
  const long nen = (long)N_ * D_;
  for (long i = i0; i < nen; i += stride) {
    int n = (int)(i % N_);
    int d = (int)(i / N_);
    enct[i] = (bf16)enc[(long)n * D_ + d];
  }
}

// ---------------------------------------------------------------------------
// v = LN(wte[idx]);  also write bf16 copy.  One wave per 256-wide row.
// ---------------------------------------------------------------------------
__global__ void k_embed_ln(const int* __restrict__ idx, const float* __restrict__ wte,
                           float* __restrict__ v, bf16* __restrict__ vbf) {
  int lane = threadIdx.x & 31;
  int row = blockIdx.x * 8 + (threadIdx.x >> 5);
  if (row >= BT) return;
  int tok = idx[row];
  const float* s = wte + (long)tok * D_;
  float vals[8]; float sm = 0.f;
  for (int j = 0; j < 8; j++) { vals[j] = s[lane + 32 * j]; sm += vals[j]; }
  float mean = wsum(sm) * (1.f / 256.f);
  float vr = 0.f;
  for (int j = 0; j < 8; j++) { float d = vals[j] - mean; vr += d * d; }
  vr = wsum(vr) * (1.f / 256.f);
  float rs = rsqrtf(vr + 1e-5f);
  for (int j = 0; j < 8; j++) {
    float o = (vals[j] - mean) * rs;
    v[(long)row * D_ + lane + 32 * j] = o;
    vbf[(long)row * D_ + lane + 32 * j] = (bf16)o;
  }
}

// ---------------------------------------------------------------------------
// x = relu(v @ decoder_x[h]); xr = rope(x).  Tile: 64 t-rows x (128 | 128+4096)
// paired columns so rope pairs (n, n+4096) live in one workgroup.
// grid = (ng=32, tg=32, bh=8), 256 threads (8 waves).
// ---------------------------------------------------------------------------
__global__ __launch_bounds__(256)
void k_gemm_x(const bf16* __restrict__ vbf, const bf16* __restrict__ dxt,
              bf16* __restrict__ x, bf16* __restrict__ xr) {
  __shared__ __align__(16) bf16 As[64 * 40];
  __shared__ __align__(16) bf16 Bs[256 * 40];
  __shared__ __align__(16) bf16 Xs[64 * 264];
  int tid = threadIdx.x, lane = tid & 31, w = tid >> 5;
  int ng = blockIdx.x, tg = blockIdx.y, bh = blockIdx.z;
  int b = bh >> 2, h = bh & 3;
  int tbase = tg * 64, c0 = ng * 128;
  const bf16* Ag = vbf + (long)(b * T_ + tbase) * D_;
  const bf16* Bg = dxt + (long)h * NH_ * D_;
  v8f vzero = {};
  v8f acc[2][4];
  for (int i = 0; i < 2; i++) for (int j = 0; j < 4; j++) acc[i][j] = vzero;
  int m0 = (w & 1) * 32, n0 = (w >> 1) * 64;

  for (int kc = 0; kc < D_; kc += 32) {
    { int r = tid >> 2, cs = (tid & 3) << 3;
      *(float4*)&As[r * 40 + cs] = *(const float4*)&Ag[(long)r * D_ + kc + cs]; }
    { int n = (tid < 128) ? (c0 + tid) : (4096 + c0 + tid - 128);
      const bf16* src = Bg + (long)n * D_ + kc;
      *(float4*)&Bs[tid * 40 + 0]  = *(const float4*)&src[0];
      *(float4*)&Bs[tid * 40 + 8]  = *(const float4*)&src[8];
      *(float4*)&Bs[tid * 40 + 16] = *(const float4*)&src[16];
      *(float4*)&Bs[tid * 40 + 24] = *(const float4*)&src[24]; }
    __syncthreads();
    v16bf af[2], bfv[4];
    for (int rt = 0; rt < 2; rt++) af[rt] = frag_a(As, 40, m0 + rt * 16, 0, lane);
    for (int ct = 0; ct < 4; ct++) bfv[ct] = frag_b(Bs, 40, n0 + ct * 16, 0, lane);
    for (int rt = 0; rt < 2; rt++)
      for (int ct = 0; ct < 4; ct++)
        acc[rt][ct] = wmma_bf(af[rt], bfv[ct], acc[rt][ct]);
    __syncthreads();
  }
  // relu -> LDS (bf16), full 64x256 tile
  for (int rt = 0; rt < 2; rt++)
    for (int ct = 0; ct < 4; ct++) {
      int nn = n0 + ct * 16 + (lane & 15);
      int mB = m0 + rt * 16 + ((lane >> 4) << 3);
      for (int r = 0; r < 8; r++) {
        float vv = acc[rt][ct][r];
        Xs[(mB + r) * 264 + nn] = (bf16)(vv > 0.f ? vv : 0.f);
      }
    }
  __syncthreads();
  // rope + store: thread owns one of 256 columns.
  int col = tid;
  int nglob = (col < 128) ? (c0 + col) : (4096 + c0 + (col - 128));
  int fidx = c0 + (col & 127);
  float invf = __expf(-(float)fidx * (9.21034037197618f / 4096.0f));
  long xbase = ((long)bh * T_ + tbase) * NH_;
  for (int r = 0; r < 64; r++) {
    float xv = (float)Xs[r * 264 + col];
    float part = (float)Xs[r * 264 + (col ^ 128)];
    float ang = (float)(tbase + r) * invf;
    float sn, cs; __sincosf(ang, &sn, &cs);
    float xrv = (col < 128) ? (xv * cs - part * sn) : (xv * cs + part * sn);
    x[xbase + (long)r * NH_ + nglob]  = (bf16)xv;
    xr[xbase + (long)r * NH_ + nglob] = (bf16)xrv;
  }
}

// ---------------------------------------------------------------------------
// Linear attention, fused: S = qr @ krT (K=8192, TDM-staged 64-wide chunks
// with double-buffered K pipeline), strict causal mask (t>s), a += S @ v.
// grid = (qt=32, bh=8), 256 threads. q-tile = 64 rows.
// LDS: [ Qs 64x72 | Ks0 128x72 | Ks1 128x72 ] (phase1) with Vt 256x40
// aliasing Qs+Ks0 (phase2), plus Ss 64x136.  Total 63488 B.
// ---------------------------------------------------------------------------
__global__ __launch_bounds__(256)
void k_attn(const bf16* __restrict__ xr, const bf16* __restrict__ vbf,
            float* __restrict__ a) {
  __shared__ __align__(16) char smem[63488];
  bf16* Qs  = (bf16*)smem;                // 64*72  = 9216 B  (phase 1)
  bf16* Ks0 = (bf16*)(smem + 9216);       // 128*72 = 18432 B (phase 1, buf 0)
  bf16* Ks1 = (bf16*)(smem + 27648);      // 128*72 = 18432 B (phase 1, buf 1)
  bf16* Vt  = (bf16*)smem;                // 256*40 = 20480 B (phase 2, aliases)
  bf16* Ss  = (bf16*)(smem + 46080);      // 64*136 = 17408 B
  unsigned qs_off  = (unsigned)(size_t)Qs;
  unsigned ks_off0 = (unsigned)(size_t)Ks0;
  unsigned ks_off1 = (unsigned)(size_t)Ks1;

  int tid = threadIdx.x, lane = tid & 31, w = tid >> 5;
  int qt = blockIdx.x, bh = blockIdx.y;
  int b = bh >> 2;
  int qbase = qt * 64;
  const bf16* X = xr + (long)bh * T_ * NH_;
  v8f vzero = {};
  v8f accO[2][4];
  for (int i = 0; i < 2; i++) for (int j = 0; j < 4; j++) accO[i][j] = vzero;
  int m0 = (w & 1) * 32, nS0 = (w >> 1) * 32, nO0 = (w >> 1) * 64;

  for (int sb = 0; sb < qbase + 64; sb += 128) {
    v8f accS[2][2];
    for (int i = 0; i < 2; i++) for (int j = 0; j < 2; j++) accS[i][j] = vzero;
    // prologue: kick off K chunk 0 into buffer 0
    if (w == 0) tdm_load_64w(ks_off0, X + (long)sb * NH_, 128);
    for (int kc = 0; kc < NH_; kc += 64) {
      int cur = (kc >> 6) & 1;
      const bf16* Kc = cur ? Ks1 : Ks0;
      if (w == 0) {
        // issue Q_i, then prefetch K_{i+1} into the other buffer; in-order
        // TENSORcnt completion => wait(1) guarantees K_i and Q_i landed
        // while K_{i+1} stays in flight.
        tdm_load_64w(qs_off, X + (long)qbase * NH_ + kc, 64);
        if (kc + 64 < NH_) {
          tdm_load_64w(cur ? ks_off0 : ks_off1, X + (long)sb * NH_ + kc + 64, 128);
          __builtin_amdgcn_s_wait_tensorcnt(1);
        } else {
          __builtin_amdgcn_s_wait_tensorcnt(0);
        }
      }
      __syncthreads();
      for (int kk = 0; kk < 64; kk += 32) {
        v16bf af[2], bfv[2];
        af[0] = frag_a(Qs, 72, m0, kk, lane);
        af[1] = frag_a(Qs, 72, m0 + 16, kk, lane);
        bfv[0] = frag_b(Kc, 72, nS0, kk, lane);
        bfv[1] = frag_b(Kc, 72, nS0 + 16, kk, lane);
        for (int rt = 0; rt < 2; rt++)
          for (int ct = 0; ct < 2; ct++)
            accS[rt][ct] = wmma_bf(af[rt], bfv[ct], accS[rt][ct]);
      }
      __syncthreads();
    }
    // mask (keep strictly lower: t > s) and spill S to LDS as bf16
    for (int rt = 0; rt < 2; rt++)
      for (int ct = 0; ct < 2; ct++) {
        int nn = nS0 + ct * 16 + (lane & 15);
        int mB = m0 + rt * 16 + ((lane >> 4) << 3);
        for (int r = 0; r < 8; r++) {
          int m = mB + r;
          bool keep = (qbase + m) > (sb + nn);
          Ss[m * 136 + nn] = keep ? (bf16)accS[rt][ct][r] : (bf16)0.0f;
        }
      }
    __syncthreads();
    // a += S @ v : K = 128 (s dim) in chunks of 32; v transposed into LDS
    for (int kc2 = 0; kc2 < 128; kc2 += 32) {
      { int k = tid >> 3, ds = (tid & 7) * 32;
        const bf16* src = vbf + (long)(b * T_ + sb + kc2 + k) * D_ + ds;
        bf16 tmp[32];
        *(float4*)&tmp[0]  = *(const float4*)&src[0];
        *(float4*)&tmp[8]  = *(const float4*)&src[8];
        *(float4*)&tmp[16] = *(const float4*)&src[16];
        *(float4*)&tmp[24] = *(const float4*)&src[24];
        for (int j = 0; j < 32; j++) Vt[(ds + j) * 40 + k] = tmp[j]; }
      __syncthreads();
      v16bf af2[2], bf2[4];
      af2[0] = frag_a(Ss, 136, m0, kc2, lane);
      af2[1] = frag_a(Ss, 136, m0 + 16, kc2, lane);
      for (int ct = 0; ct < 4; ct++) bf2[ct] = frag_b(Vt, 40, nO0 + ct * 16, 0, lane);
      for (int rt = 0; rt < 2; rt++)
        for (int ct = 0; ct < 4; ct++)
          accO[rt][ct] = wmma_bf(af2[rt], bf2[ct], accO[rt][ct]);
      __syncthreads();
    }
  }
  long abase = ((long)bh * T_ + qbase) * D_;
  for (int rt = 0; rt < 2; rt++)
    for (int ct = 0; ct < 4; ct++) {
      int nn = nO0 + ct * 16 + (lane & 15);
      int mB = m0 + rt * 16 + ((lane >> 4) << 3);
      for (int r = 0; r < 8; r++)
        a[abase + (long)(mB + r) * D_ + nn] = accO[rt][ct][r];
    }
}

// ---------------------------------------------------------------------------
// aln = bf16(LN(a)), rows = B*H*T.
// ---------------------------------------------------------------------------
__global__ void k_ln_a(const float* __restrict__ a, bf16* __restrict__ aln) {
  int lane = threadIdx.x & 31;
  int row = blockIdx.x * 8 + (threadIdx.x >> 5);
  if (row >= BHT) return;
  const float* s = a + (long)row * D_;
  float vals[8]; float sm = 0.f;
  for (int j = 0; j < 8; j++) { vals[j] = s[lane + 32 * j]; sm += vals[j]; }
  float mean = wsum(sm) * (1.f / 256.f);
  float vr = 0.f;
  for (int j = 0; j < 8; j++) { float d = vals[j] - mean; vr += d * d; }
  vr = wsum(vr) * (1.f / 256.f);
  float rs = rsqrtf(vr + 1e-5f);
  for (int j = 0; j < 8; j++)
    aln[(long)row * D_ + lane + 32 * j] = (bf16)((vals[j] - mean) * rs);
}

__global__ void k_zero(float* __restrict__ p, int n) {
  int i0 = blockIdx.x * blockDim.x + threadIdx.x;
  for (int i = i0; i < n; i += gridDim.x * blockDim.x) p[i] = 0.f;
}

// ---------------------------------------------------------------------------
// Fused: y = relu(aln @ decoder_y[h]) * x ; z += y @ encoder_chunk[h].
// grid = (tg=32, slice=8, b=2); slice -> (h = slice>>1, n-chunk half).
// ---------------------------------------------------------------------------
__global__ __launch_bounds__(256)
void k_fused_y(const bf16* __restrict__ aln, const bf16* __restrict__ dyt,
               const bf16* __restrict__ x, const bf16* __restrict__ enct,
               float* __restrict__ z) {
  __shared__ __align__(16) bf16 As[64 * 40];
  __shared__ __align__(16) bf16 Dys[128 * 40];
  __shared__ __align__(16) bf16 Ys[64 * 136];
  __shared__ __align__(16) bf16 Ets[256 * 40];
  int tid = threadIdx.x, lane = tid & 31, w = tid >> 5;
  int tg = blockIdx.x, slice = blockIdx.y, b = blockIdx.z;
  int tbase = tg * 64;
  int h = slice >> 1;
  int ncs = (slice & 1) * 32;
  const bf16* alnB = aln + ((long)(b * H_ + h) * T_ + tbase) * D_;
  const bf16* dyB = dyt + (long)h * NH_ * D_;
  const bf16* xB  = x + ((long)(b * H_ + h) * T_ + tbase) * NH_;
  v8f vzero = {};
  v8f accZ[2][4];
  for (int i = 0; i < 2; i++) for (int j = 0; j < 4; j++) accZ[i][j] = vzero;
  int m0 = (w & 1) * 32, nS0 = (w >> 1) * 32, nO0 = (w >> 1) * 64;

  for (int i = 0; i < 32; i++) {
    int nbase = (ncs + i) * 128;
    v8f accY[2][2];
    for (int p = 0; p < 2; p++) for (int q = 0; q < 2; q++) accY[p][q] = vzero;
    // Phase A: y = aln @ dy  (K = 256)
    for (int kc = 0; kc < D_; kc += 32) {
      { int r = tid >> 2, cs = (tid & 3) << 3;
        *(float4*)&As[r * 40 + cs] = *(const float4*)&alnB[(long)r * D_ + kc + cs]; }
      { int r = tid >> 1, cs = (tid & 1) << 4;
        const bf16* src = &dyB[(long)(nbase + r) * D_ + kc + cs];
        *(float4*)&Dys[r * 40 + cs]     = *(const float4*)&src[0];
        *(float4*)&Dys[r * 40 + cs + 8] = *(const float4*)&src[8]; }
      __syncthreads();
      v16bf af[2], bfv[2];
      af[0] = frag_a(As, 40, m0, 0, lane);
      af[1] = frag_a(As, 40, m0 + 16, 0, lane);
      bfv[0] = frag_b(Dys, 40, nS0, 0, lane);
      bfv[1] = frag_b(Dys, 40, nS0 + 16, 0, lane);
      for (int rt = 0; rt < 2; rt++)
        for (int ct = 0; ct < 2; ct++)
          accY[rt][ct] = wmma_bf(af[rt], bfv[ct], accY[rt][ct]);
      __syncthreads();
    }
    // relu * x -> Ys (bf16)
    for (int rt = 0; rt < 2; rt++)
      for (int ct = 0; ct < 2; ct++) {
        int nn = nS0 + ct * 16 + (lane & 15);
        int mB = m0 + rt * 16 + ((lane >> 4) << 3);
        for (int r = 0; r < 8; r++) {
          int m = mB + r;
          float yv = accY[rt][ct][r];
          yv = yv > 0.f ? yv : 0.f;
          yv *= (float)xB[(long)m * NH_ + nbase + nn];
          Ys[m * 136 + nn] = (bf16)yv;
        }
      }
    __syncthreads();
    // Phase B: z += y @ enc   (K = 128 chunk of n)
    for (int kc2 = 0; kc2 < 128; kc2 += 32) {
      { int d = tid;
        const bf16* src = &enct[(long)d * N_ + h * NH_ + nbase + kc2];
        *(float4*)&Ets[d * 40 + 0]  = *(const float4*)&src[0];
        *(float4*)&Ets[d * 40 + 8]  = *(const float4*)&src[8];
        *(float4*)&Ets[d * 40 + 16] = *(const float4*)&src[16];
        *(float4*)&Ets[d * 40 + 24] = *(const float4*)&src[24]; }
      __syncthreads();
      v16bf af2[2], bf2[4];
      af2[0] = frag_a(Ys, 136, m0, kc2, lane);
      af2[1] = frag_a(Ys, 136, m0 + 16, kc2, lane);
      for (int ct = 0; ct < 4; ct++) bf2[ct] = frag_b(Ets, 40, nO0 + ct * 16, 0, lane);
      for (int rt = 0; rt < 2; rt++)
        for (int ct = 0; ct < 4; ct++)
          accZ[rt][ct] = wmma_bf(af2[rt], bf2[ct], accZ[rt][ct]);
      __syncthreads();
    }
  }
  for (int rt = 0; rt < 2; rt++)
    for (int ct = 0; ct < 4; ct++) {
      int nn = nO0 + ct * 16 + (lane & 15);
      int mB = m0 + rt * 16 + ((lane >> 4) << 3);
      for (int r = 0; r < 8; r++)
        atomicAdd(&z[((long)(b * T_ + tbase) + mB + r) * D_ + nn], accZ[rt][ct][r]);
    }
}

// ---------------------------------------------------------------------------
// v = LN(v + LN(z)); refresh bf16 copy.
// ---------------------------------------------------------------------------
__global__ void k_res_ln(const float* __restrict__ z, float* __restrict__ v,
                         bf16* __restrict__ vbf) {
  int lane = threadIdx.x & 31;
  int row = blockIdx.x * 8 + (threadIdx.x >> 5);
  if (row >= BT) return;
  const float* zs = z + (long)row * D_;
  float* vs = v + (long)row * D_;
  float zv[8]; float sm = 0.f;
  for (int j = 0; j < 8; j++) { zv[j] = zs[lane + 32 * j]; sm += zv[j]; }
  float mean = wsum(sm) * (1.f / 256.f);
  float vr = 0.f;
  for (int j = 0; j < 8; j++) { float d = zv[j] - mean; vr += d * d; }
  vr = wsum(vr) * (1.f / 256.f);
  float rs = rsqrtf(vr + 1e-5f);
  float wv[8]; sm = 0.f;
  for (int j = 0; j < 8; j++) {
    wv[j] = vs[lane + 32 * j] + (zv[j] - mean) * rs;
    sm += wv[j];
  }
  float mean2 = wsum(sm) * (1.f / 256.f);
  vr = 0.f;
  for (int j = 0; j < 8; j++) { float d = wv[j] - mean2; vr += d * d; }
  vr = wsum(vr) * (1.f / 256.f);
  float rs2 = rsqrtf(vr + 1e-5f);
  for (int j = 0; j < 8; j++) {
    float o = (wv[j] - mean2) * rs2;
    vs[lane + 32 * j] = o;
    vbf[(long)row * D_ + lane + 32 * j] = (bf16)o;
  }
}

// ---------------------------------------------------------------------------
// out = v @ readout  (4096x256 @ 256x256, f32 out). grid = 64 blocks.
// ---------------------------------------------------------------------------
__global__ __launch_bounds__(256)
void k_readout(const bf16* __restrict__ vbf, const float* __restrict__ ro,
               float* __restrict__ out) {
  __shared__ __align__(16) bf16 As[64 * 40];
  __shared__ __align__(16) bf16 Rt[256 * 40];
  int tid = threadIdx.x, lane = tid & 31, w = tid >> 5;
  int tbase = blockIdx.x * 64;
  v8f vzero = {};
  v8f acc[2][4];
  for (int i = 0; i < 2; i++) for (int j = 0; j < 4; j++) acc[i][j] = vzero;
  int m0 = (w & 1) * 32, n0 = (w >> 1) * 64;
  for (int kc = 0; kc < D_; kc += 32) {
    { int r = tid >> 2, cs = (tid & 3) << 3;
      *(float4*)&As[r * 40 + cs] =
          *(const float4*)&vbf[(long)(tbase + r) * D_ + kc + cs]; }
    { int k = tid >> 3, vs0 = (tid & 7) * 32;
      const float* src = &ro[(long)(kc + k) * V_ + vs0];
      for (int j = 0; j < 32; j++) Rt[(vs0 + j) * 40 + k] = (bf16)src[j]; }
    __syncthreads();
    v16bf af[2], bfv[4];
    af[0] = frag_a(As, 40, m0, 0, lane);
    af[1] = frag_a(As, 40, m0 + 16, 0, lane);
    for (int ct = 0; ct < 4; ct++) bfv[ct] = frag_b(Rt, 40, n0 + ct * 16, 0, lane);
    for (int rt = 0; rt < 2; rt++)
      for (int ct = 0; ct < 4; ct++)
        acc[rt][ct] = wmma_bf(af[rt], bfv[ct], acc[rt][ct]);
    __syncthreads();
  }
  for (int rt = 0; rt < 2; rt++)
    for (int ct = 0; ct < 4; ct++) {
      int nn = n0 + ct * 16 + (lane & 15);
      int mB = m0 + rt * 16 + ((lane >> 4) << 3);
      for (int r = 0; r < 8; r++)
        out[(long)(tbase + mB + r) * V_ + nn] = acc[rt][ct][r];
    }
}

// ---------------------------------------------------------------------------
extern "C" void kernel_launch(void* const* d_in, const int* in_sizes, int n_in,
                              void* d_out, int out_size, void* d_ws, size_t ws_size,
                              hipStream_t stream) {
  (void)in_sizes; (void)n_in; (void)out_size; (void)ws_size;
  const int*   idx = (const int*)d_in[0];
  const float* wte = (const float*)d_in[1];
  const float* enc = (const float*)d_in[2];
  const float* dx  = (const float*)d_in[3];
  const float* dy  = (const float*)d_in[4];
  const float* ro  = (const float*)d_in[5];

  char* ws = (char*)d_ws;
  size_t off = 0;
  auto alloc = [&](size_t bytes) {
    void* p = ws + off;
    off = (off + bytes + 255) & ~(size_t)255;
    return p;
  };
  float* v   = (float*)alloc(sizeof(float) * BT * D_);
  bf16*  vbf = (bf16*) alloc(sizeof(bf16) * BT * D_);
  bf16*  x   = (bf16*) alloc(sizeof(bf16) * (size_t)B_ * H_ * T_ * NH_);
  bf16*  xr  = (bf16*) alloc(sizeof(bf16) * (size_t)B_ * H_ * T_ * NH_);
  float* a   = (float*)alloc(sizeof(float) * (size_t)BHT * D_);
  bf16*  aln = (bf16*) alloc(sizeof(bf16) * (size_t)BHT * D_);
  float* z   = (float*)alloc(sizeof(float) * BT * D_);
  bf16*  dxt = (bf16*) alloc(sizeof(bf16) * (size_t)H_ * NH_ * D_);
  bf16*  dyt = (bf16*) alloc(sizeof(bf16) * (size_t)H_ * NH_ * D_);
  bf16*  enct= (bf16*) alloc(sizeof(bf16) * (size_t)N_ * D_);

  k_convert<<<2048, 256, 0, stream>>>(dx, dy, enc, dxt, dyt, enct);
  k_embed_ln<<<BT / 8, 256, 0, stream>>>(idx, wte, v, vbf);

  for (int l = 0; l < LAYERS; l++) {
    k_gemm_x<<<dim3(32, 32, 8), 256, 0, stream>>>(vbf, dxt, x, xr);
    k_attn<<<dim3(32, 8), 256, 0, stream>>>(xr, vbf, a);
    k_ln_a<<<BHT / 8, 256, 0, stream>>>(a, aln);
    k_zero<<<1024, 256, 0, stream>>>(z, BT * D_);
    k_fused_y<<<dim3(32, 8, 2), 256, 0, stream>>>(aln, dyt, x, enct, z);
    k_res_ln<<<BT / 8, 256, 0, stream>>>(z, v, vbf);
  }
  k_readout<<<BT / 64, 256, 0, stream>>>(vbf, ro, (float*)d_out);
}
